// NetworkAction_86131274154571
// MI455X (gfx1250) — matrix operand order repaired and checked by reading, not compiled, not run on hardware
//
#include <hip/hip_runtime.h>
#include <hip/hip_bf16.h>
#include <math.h>

// ---------------------------------------------------------------------------
// GNN barrier-policy net, fused for MI455X (gfx1250, wave32, WMMA).
//   Kernel A: pairwise conv1(5->64) -> conv2(64->128) -> mask -> maxpool(j)
//             via v_wmma_f32_16x16x32_f16, fully fused in LDS.
//             conv1 bias folded into the matmul (constant-1 feature column,
//             C = inline 0); conv2 bias in hoisted C fragments; mask+pool via
//             -inf select + max tree (relu folded into the >=0 pool floor).
//             conv1 issues its 4 WMMAs back-to-back to cover hazard windows.
//   Kernel B: per-agent MLP head (0.2% of FLOPs, VALU).
// ---------------------------------------------------------------------------

typedef __attribute__((ext_vector_type(16))) _Float16 v16h;
typedef __attribute__((ext_vector_type(8)))  _Float16 v8h;
typedef __attribute__((ext_vector_type(8)))  float    v8f;

union AFrag { v16h v; v8h h[2]; };

#define NAG   1024
#define XROW  40   // padded halves per feature row (K=32 used; stride 20 dwords -> conflict-free)
#define HROW  72   // padded halves per h1 row (K=64 used; stride 36 dwords -> conflict-free)
#define NEGBIG (-3.402823466e38f)

// --------------------------- pooled init -----------------------------------
__global__ __launch_bounds__(512) void zero_pooled_kernel(float* __restrict__ pooled) {
    int idx = blockIdx.x * blockDim.x + threadIdx.x;
    if (idx < NAG * 128) pooled[idx] = 0.0f;
}

// --------------------------- fused conv+pool -------------------------------
// grid: (64, 8)  -> blockIdx.x = i-tile (16 agents), blockIdx.y = j-chunk (8 j-tiles)
// block: 128 threads = 4 waves (wave32)
__global__ __launch_bounds__(128) void convpool_kernel(
    const float* __restrict__ s,
    const float* __restrict__ w1, const float* __restrict__ b1,   // (64,5), (64)
    const float* __restrict__ w2, const float* __restrict__ b2,   // (128,64), (128)
    float* __restrict__ pooled)                                   // (1024,128) in d_ws
{
    __shared__ _Float16 xbuf[256 * XROW];   // pair features, K padded 5->32 (col5 = 1.0 bias feature)
    __shared__ _Float16 h1buf[256 * HROW];  // relu(conv1) activations, f16
    __shared__ float    d2buf[256];         // dist^2 per pair

    const int tid  = threadIdx.x;
    const int lane = tid & 31;
    const int wave = tid >> 5;      // 0..3
    const int n15  = lane & 15;
    const int kh   = lane >> 4;     // lane half: 0 or 1
    const int blk  = blockIdx.x;    // 0..63 : i-tile

    // one-time zero of xbuf (padding columns never rewritten afterwards)
    {
        uint4 z = {0u, 0u, 0u, 0u};
        uint4* xb4 = (uint4*)xbuf;
        for (int idx = tid; idx < (256 * XROW * 2) / 16; idx += 128) xb4[idx] = z;
    }

    // ---- conv1 B fragments (B is 32x16, K x N); bias folded in at K=5 ----
    // B layout: lane n = lane&15, lanes 0-15 hold K=0..7,16..23, lanes 16-31 hold K=8..15,24..31
    v16h b1f[4];
    #pragma unroll
    for (int nt = 0; nt < 4; ++nt) {
        #pragma unroll
        for (int e = 0; e < 16; ++e) {
            float val = 0.0f;
            if (e < 5)       val = (kh == 0) ? w1[(nt * 16 + n15) * 5 + e] : 0.0f; // K=e (kh==0)
            else if (e == 5) val = (kh == 0) ? b1[nt * 16 + n15] : 0.0f;           // bias row K=5
            b1f[nt][e] = (_Float16)val;
        }
    }

    // ---- conv2 B fragments + hoisted biased C fragments (owned tiles {wave, wave+4}) ----
    v16h b2f[2][2];
    v8f  cbias[2];
    #pragma unroll
    for (int u = 0; u < 2; ++u) {
        int nt = wave + 4 * u;
        #pragma unroll
        for (int ks = 0; ks < 2; ++ks)
            #pragma unroll
            for (int e = 0; e < 16; ++e) {
                int K = ks * 32 + ((e < 8) ? (kh * 8 + e) : (16 + kh * 8 + (e - 8)));
                b2f[u][ks][e] = (_Float16)w2[(nt * 16 + n15) * 64 + K];
            }
        float bv = b2[nt * 16 + n15];
        #pragma unroll
        for (int r = 0; r < 8; ++r) cbias[u][r] = bv;
    }

    // running max-pool accumulators: [owned n-tile][i_local], valid in lanes 0..15 (ch = lane)
    float runmax[2][16];
    #pragma unroll
    for (int u = 0; u < 2; ++u)
        #pragma unroll
        for (int t = 0; t < 16; ++t) runmax[u][t] = 0.0f;

    // this thread owns pairs p0, p0+1 (same i_local, consecutive j_local)
    const int p0    = 2 * tid;
    const int iloc  = p0 >> 4;
    const int iglob = blk * 16 + iloc;
    const float4 si = ((const float4*)s)[iglob];

    __syncthreads();  // xbuf zeros visible

    // constant-1 bias feature column (K=5) for this thread's two pair rows
    xbuf[(p0    ) * XROW + 5] = (_Float16)1.0f;
    xbuf[(p0 + 1) * XROW + 5] = (_Float16)1.0f;

    const int jt0 = blockIdx.y * 8;
    for (int jt = jt0; jt < jt0 + 8; ++jt) {
        // ---------- phase 1: pair features -> LDS (wave-local rows) ----------
        #pragma unroll
        for (int pp = 0; pp < 2; ++pp) {
            int p  = p0 + pp;
            int jg = jt * 16 + (p & 15);
            float4 sj = ((const float4*)s)[jg];
            float dx = si.x - sj.x, dy = si.y - sj.y;
            float dz = si.z - sj.z, dw = si.w - sj.w;
            _Float16* xr = &xbuf[p * XROW];
            xr[0] = (_Float16)dx;  xr[1] = (_Float16)dy;
            xr[2] = (_Float16)dz;  xr[3] = (_Float16)dw;
            xr[4] = (_Float16)((iglob == jg) ? 1.0f : 0.0f);   // eye channel
            d2buf[p] = dx * dx + dy * dy;
        }
        // NO barrier here: conv1 tile t = wave*4+tt reads only xbuf rows
        // 64*wave..64*wave+63, which this wave's own threads just wrote;
        // LDS ops are in-order per wave (DScnt), compiler emits the wait.
        // d2buf is only read cross-wave in conv2, after the barrier below.

        // ---------- conv1 (K padded to 32, bias in K=5 row, C = 0) ----------
        // A 16-bit layout: lane row m=lane&15; elems e<8 -> K=kh*8+e, e>=8 -> K=16+kh*8+(e-8)
        #pragma unroll
        for (int tt = 0; tt < 4; ++tt) {
            int t = wave * 4 + tt;
            AFrag a;
            const v8h* xp = (const v8h*)&xbuf[(t * 16 + n15) * XROW];
            a.h[0] = xp[kh];
            a.h[1] = xp[2 + kh];
            // issue all 4 WMMAs back-to-back (independent D), then consume:
            // each covers the others' D->VALU hazard window (no v_nop stalls)
            v8f d[4];
            #pragma unroll
            for (int nt = 0; nt < 4; ++nt) {
                v8f cz = {};   // inline-0 accumulator (bias comes through the matmul)
                d[nt] = __builtin_amdgcn_wmma_f32_16x16x32_f16(false, a.v, false, b1f[nt],
                                                               (short)0, cz, false, false);
            }
            // D layout: lane holds N=lane&15, M = r + 8*kh ; relu via med3, store f16
            #pragma unroll
            for (int nt = 0; nt < 4; ++nt)
                #pragma unroll
                for (int r = 0; r < 8; ++r) {
                    int p = t * 16 + r + 8 * kh;
                    float hv = __builtin_amdgcn_fmed3f(d[nt][r], 0.0f, 3.0e38f);  // relu, 1 instr
                    h1buf[p * HROW + nt * 16 + n15] = (_Float16)hv;
                }
        }
        __syncthreads();   // h1buf (and d2buf) now visible to all waves

        // ---------- conv2 + mask + running max-pool ----------
        #pragma unroll
        for (int t = 0; t < 16; ++t) {
            AFrag a0, a1;
            const v8h* hp = (const v8h*)&h1buf[(t * 16 + n15) * HROW];
            a0.h[0] = hp[kh];      a0.h[1] = hp[2 + kh];   // K = 0..31
            a1.h[0] = hp[4 + kh];  a1.h[1] = hp[6 + kh];   // K = 32..63
            const float4* dp = (const float4*)&d2buf[t * 16 + 8 * kh];
            float4 dA = dp[0], dB = dp[1];
            float d2a[8] = {dA.x, dA.y, dA.z, dA.w, dB.x, dB.y, dB.z, dB.w};

            // two independent accumulator chains, interleaved to space WMMA hazards
            v8f c0 = __builtin_amdgcn_wmma_f32_16x16x32_f16(false, a0.v, false, b2f[0][0],
                                                            (short)0, cbias[0], false, false);
            v8f c1 = __builtin_amdgcn_wmma_f32_16x16x32_f16(false, a0.v, false, b2f[1][0],
                                                            (short)0, cbias[1], false, false);
            c0 = __builtin_amdgcn_wmma_f32_16x16x32_f16(false, a1.v, false, b2f[0][1],
                                                        (short)0, c0, false, false);
            c1 = __builtin_amdgcn_wmma_f32_16x16x32_f16(false, a1.v, false, b2f[1][1],
                                                        (short)0, c1, false, false);

            // mask (dist^2 < 0.25) selects -inf for dropped pairs; relu folded into the
            // >=0 floor of runmax/atomicMax (reference max always includes 0 terms).
            float v0 = NEGBIG, v1 = NEGBIG;
            #pragma unroll
            for (int r = 0; r < 8; ++r) {
                bool m = d2a[r] < 0.25f;                 // shared compare for both chains
                v0 = fmaxf(v0, m ? c0[r] : NEGBIG);
                v1 = fmaxf(v1, m ? c1[r] : NEGBIG);
            }
            v0 = fmaxf(v0, __shfl_xor(v0, 16));          // fold lane halves (M 0..7 vs 8..15)
            v1 = fmaxf(v1, __shfl_xor(v1, 16));
            runmax[0][t] = fmaxf(runmax[0][t], v0);      // runmax >= 0 provides the relu floor
            runmax[1][t] = fmaxf(runmax[1][t], v1);
        }
        __syncthreads();   // protect h1buf/d2buf before next iteration overwrites
    }

    // merge the 8 j-chunks: all published values >= 0, so int atomicMax on bits is exact
    if (kh == 0) {
        #pragma unroll
        for (int u = 0; u < 2; ++u)
            #pragma unroll
            for (int t = 0; t < 16; ++t) {
                int* dst = (int*)&pooled[(blk * 16 + t) * 128 + (wave + 4 * u) * 16 + n15];
                atomicMax(dst, __float_as_int(runmax[u][t]));
            }
    }
}

// --------------------------- per-agent MLP head ----------------------------
__global__ __launch_bounds__(256) void mlp_kernel(
    const float* __restrict__ s, const float* __restrict__ g,
    const float* __restrict__ pooled,
    const float* __restrict__ w1, const float* __restrict__ b1,   // (64,132)
    const float* __restrict__ w2, const float* __restrict__ b2,   // (128,64)
    const float* __restrict__ w3, const float* __restrict__ b3,   // (64,128)
    const float* __restrict__ w4, const float* __restrict__ b4,   // (4,64)
    float* __restrict__ out)                                      // (1024,2)
{
    int i = blockIdx.x * blockDim.x + threadIdx.x;
    if (i >= NAG) return;

    float feat[132];
    for (int c = 0; c < 128; ++c) feat[c] = pooled[i * 128 + c];
    float sgx = s[i * 4 + 0] - g[i * 2 + 0];
    float sgy = s[i * 4 + 1] - g[i * 2 + 1];
    float vx  = s[i * 4 + 2];
    float vy  = s[i * 4 + 3];
    feat[128] = sgx; feat[129] = sgy; feat[130] = vx; feat[131] = vy;

    float z1[64];
    for (int o = 0; o < 64; ++o) {
        float acc = b1[o];
        for (int c = 0; c < 132; ++c) acc = fmaf(feat[c], w1[o * 132 + c], acc);
        z1[o] = fmaxf(acc, 0.0f);
    }
    float z2[128];
    for (int o = 0; o < 128; ++o) {
        float acc = b2[o];
        for (int c = 0; c < 64; ++c) acc = fmaf(z1[c], w2[o * 64 + c], acc);
        z2[o] = fmaxf(acc, 0.0f);
    }
    float z3[64];
    for (int o = 0; o < 64; ++o) {
        float acc = b3[o];
        for (int c = 0; c < 128; ++c) acc = fmaf(z2[c], w3[o * 128 + c], acc);
        z3[o] = fmaxf(acc, 0.0f);
    }
    float kk[4];
    for (int o = 0; o < 4; ++o) {
        float acc = b4[o];
        for (int c = 0; c < 64; ++c) acc = fmaf(z3[c], w4[o * 64 + c], acc);
        kk[o] = 2.0f / (1.0f + __expf(-acc)) - 1.0f;   // 2*sigmoid - 1
    }
    out[i * 2 + 0] = -(kk[0] * sgx + kk[1] * vx);
    out[i * 2 + 1] = -(kk[2] * sgy + kk[3] * vy);
}

// ---------------------------------------------------------------------------
extern "C" void kernel_launch(void* const* d_in, const int* in_sizes, int n_in,
                              void* d_out, int out_size, void* d_ws, size_t ws_size,
                              hipStream_t stream) {
    const float* s   = (const float*)d_in[0];
    const float* g   = (const float*)d_in[1];
    const float* c1w = (const float*)d_in[2];
    const float* c1b = (const float*)d_in[3];
    const float* c2w = (const float*)d_in[4];
    const float* c2b = (const float*)d_in[5];
    const float* f1w = (const float*)d_in[6];
    const float* f1b = (const float*)d_in[7];
    const float* f2w = (const float*)d_in[8];
    const float* f2b = (const float*)d_in[9];
    const float* f3w = (const float*)d_in[10];
    const float* f3b = (const float*)d_in[11];
    const float* f4w = (const float*)d_in[12];
    const float* f4b = (const float*)d_in[13];

    float* pooled = (float*)d_ws;            // 1024*128 f32 = 512 KB scratch
    float* out    = (float*)d_out;

    zero_pooled_kernel<<<(NAG * 128 + 511) / 512, 512, 0, stream>>>(pooled);
    convpool_kernel<<<dim3(64, 8), 128, 0, stream>>>(s, c1w, c1b, c2w, c2b, pooled);
    mlp_kernel<<<NAG / 256, 256, 0, stream>>>(s, g, pooled,
                                              f1w, f1b, f2w, f2b, f3w, f3b, f4w, f4b, out);
}
